// IntClassifier_3521873183459
// MI455X (gfx1250) — 8-bit matrix operand order repaired and checked by reading, compiled, not run on hardware
//
#include <hip/hip_runtime.h>
#include <stdint.h>

#define B_DIM 256
#define C_DIM 2048
#define L_DIM 512
#define N_DIM 1000
#define N_PAD 1008
#define FINAL_SHIFT 4
#define FC_SHIFT 11

typedef __attribute__((ext_vector_type(8))) int v8i;

__device__ __forceinline__ int dead_shift_i32(int x, int sh) {
    int a = x < 0 ? -x : x;
    return (a >= (1 << sh)) ? (x >> sh) : 0;
}

__global__ void pack_w_kernel(const int* __restrict__ w32,
                              int* __restrict__ wp  ) {
    int idx4 = blockIdx.x * blockDim.x + threadIdx.x;
    int n   = idx4 >> 9;
    int c4  = idx4 & 511;
    int packed = 0;
    if (n < N_DIM) {
        int4 v = *(const int4*)(w32 + (size_t)n * C_DIM + (c4 << 2));
        packed = (v.x & 255) | ((v.y & 255) << 8) |
                 ((v.z & 255) << 16) | ((v.w & 255) << 24);
    }
    wp[idx4] = packed;
}

__global__ void pool_kernel(const int* __restrict__ x,
                            unsigned char* __restrict__ accLo,
                            signed char* __restrict__ accHi) {
    const int lane     = threadIdx.x & 31;
    const int rowInBlk = threadIdx.x >> 5;
    const size_t row   = (size_t)blockIdx.x * 8 + rowInBlk;
    const int* p = x + row * L_DIM;

    int s = 0;
#pragma unroll
    for (int j = 0; j < 4; ++j) {
        int4 v = *(const int4*)(p + j * 128 + lane * 4);
        s += v.x + v.y + v.z + v.w;
    }
#pragma unroll
    for (int m = 16; m >= 1; m >>= 1)
        s += __shfl_xor(s, m, 32);

    if (lane == 0) {
        int acc = dead_shift_i32(s, FINAL_SHIFT);
        accLo[row] = (unsigned char)(acc & 255);
        accHi[row] = (signed char)(acc >> 8);
    }
}

__global__ void __launch_bounds__(128)
gemm_wmma_kernel(const unsigned char* __restrict__ accLo,
                 const signed char*  __restrict__ accHi,
                 const signed char*  __restrict__ wp,
                 float* __restrict__ out) {
    const int lane = threadIdx.x & 31;
    const int wave = threadIdx.x >> 5;
    const int tile = blockIdx.x * 4 + wave;
    const int nt   = tile % 63;
    const int mt   = tile / 63;
    const int half = lane >> 4;
    const int lr   = lane & 15;

    const unsigned char* pAlo = accLo + (size_t)(mt * 16 + lr) * C_DIM + half * 8;
    const signed char*   pAhi = accHi + (size_t)(mt * 16 + lr) * C_DIM + half * 8;
    const signed char*   pB   = wp    + (size_t)(nt * 16 + lr) * C_DIM + half * 8;

    v8i cLo = {};
    v8i cHi = {};

#pragma unroll 2
    for (int k = 0; k < C_DIM; k += 64) {
        int2 l0 = *(const int2*)(pAlo + k);
        int2 l1 = *(const int2*)(pAlo + k + 16);
        int2 l2 = *(const int2*)(pAlo + k + 32);
        int2 l3 = *(const int2*)(pAlo + k + 48);
        v8i aLo = {l0.x, l0.y, l1.x, l1.y, l2.x, l2.y, l3.x, l3.y};

        int2 h0 = *(const int2*)(pAhi + k);
        int2 h1 = *(const int2*)(pAhi + k + 16);
        int2 h2 = *(const int2*)(pAhi + k + 32);
        int2 h3 = *(const int2*)(pAhi + k + 48);
        v8i aHi = {h0.x, h0.y, h1.x, h1.y, h2.x, h2.y, h3.x, h3.y};

        int2 b0 = *(const int2*)(pB + k);
        int2 b1 = *(const int2*)(pB + k + 16);
        int2 b2 = *(const int2*)(pB + k + 32);
        int2 b3 = *(const int2*)(pB + k + 48);
        v8i b = {b0.x, b0.y, b1.x, b1.y, b2.x, b2.y, b3.x, b3.y};

        cLo = __builtin_amdgcn_wmma_i32_16x16x64_iu8(false, aLo, true, b, cLo, false, false);
        cHi = __builtin_amdgcn_wmma_i32_16x16x64_iu8(true,  aHi, true, b, cHi, false, false);
    }

    const int m0 = mt * 16 + half * 8;
    const int n  = nt * 16 + lr;
    if (n < N_DIM) {
#pragma unroll
        for (int r = 0; r < 8; ++r) {
            int y = (cHi[r] << 8) + cLo[r];
            int z = dead_shift_i32(y, FC_SHIFT);
            z = z < -128 ? -128 : (z > 127 ? 127 : z);
            out[(size_t)(m0 + r) * N_DIM + n] = (float)z;
        }
    }
}

extern "C" void kernel_launch(void* const* d_in, const int* in_sizes, int n_in,
                              void* d_out, int out_size, void* d_ws, size_t ws_size,
                              hipStream_t stream) {
    (void)in_sizes; (void)n_in; (void)out_size; (void)ws_size;

    const int* x   = (const int*)d_in[0];
    const int* w32 = (const int*)d_in[1];
    float* out = (float*)d_out;

    unsigned char* ws    = (unsigned char*)d_ws;
    unsigned char* accLo = ws;
    signed char*   accHi = (signed char*)(ws + (size_t)B_DIM * C_DIM);
    signed char*   wpack = (signed char*)(ws + (size_t)2 * B_DIM * C_DIM);

    {
        int threads = (N_PAD * C_DIM) / 4;
        pack_w_kernel<<<threads / 256, 256, 0, stream>>>(w32, (int*)wpack);
    }
    {
        int rows = B_DIM * C_DIM;
        pool_kernel<<<rows / 8, 256, 0, stream>>>(x, accLo, accHi);
    }
    {
        gemm_wmma_kernel<<<(16 * 63) / 4, 128, 0, stream>>>(accLo, accHi, wpack, out);
    }
}
